// E_BertSelfAttention_44135083934329
// MI455X (gfx1250) — matrix operand order repaired and checked by reading, not compiled, not run on hardware
//
#include <hip/hip_runtime.h>

// Shapes (match reference)
#define Bq   4
#define Sq   1024
#define HIDq 768
#define Hq   12
#define Dq   64

typedef __attribute__((ext_vector_type(16))) _Float16 v16h;
typedef __attribute__((ext_vector_type(8)))  _Float16 v8h;
typedef __attribute__((ext_vector_type(8)))  float    v8f;

union AF { v16h v; _Float16 h[16]; v8h q[2]; };

// B-fragment (32x16, 16-bit): lane l holds N = l%16, K = 16*(l/16) + 0..15,
// contiguous along K per lane -> 32 contiguous bytes.
__device__ __forceinline__ v16h load_bfrag(const _Float16* p) {
    AF b;
    b.q[0] = *(const v8h*)(p);
    b.q[1] = *(const v8h*)(p + 8);
    return b.v;
}

// A-fragment (16x32, 16-bit) from a row-major f16 row pointer:
// lane l holds M = l%16; halves[0..15] = row[k0hi + 0..15]
// (k0hi = chunk + 16*(l/16)).
__device__ __forceinline__ v16h load_afrag(const _Float16* rowp, int k0hi) {
    AF a;
    a.q[0] = *(const v8h*)(rowp + k0hi);
    a.q[1] = *(const v8h*)(rowp + k0hi + 8);
    return a.v;
}

__device__ __forceinline__ v8f wmma_f16(v16h a, v16h b, v8f c) {
    return __builtin_amdgcn_wmma_f32_16x16x32_f16(false, a, false, b, (short)0, c,
                                                  false, false);
}

// ---------------------------------------------------------------------------
// Kernel 0: transpose + convert weights: wt[which][col][k] = (f16) W[k][col]
// ---------------------------------------------------------------------------
__global__ void k_wcvt(const float* __restrict__ Wq, const float* __restrict__ Wk,
                       const float* __restrict__ Wv, _Float16* __restrict__ wt) {
    int idx = blockIdx.x * blockDim.x + threadIdx.x;   // 3*768*768 total
    int which = idx / (HIDq * HIDq);
    int rem   = idx % (HIDq * HIDq);
    int c = rem / HIDq;
    int k = rem % HIDq;
    const float* W = (which == 0) ? Wq : ((which == 1) ? Wk : Wv);
    wt[idx] = (_Float16)W[k * HIDq + c];
}

// ---------------------------------------------------------------------------
// Kernel 1: fused QKV projection. One wave -> one 32x64 output macro-tile
// (2 row-tiles x 4 col-tiles, 8 WMMA accumulators -> ~16 FLOP/byte from L2).
// grid = (4096/32, 768/64, 3). Q,K stored f16 [B,H,S,D]; V stored [B,H,D,S].
// ---------------------------------------------------------------------------
__global__ __launch_bounds__(32) void k_qkv(
    const float* __restrict__ X, const _Float16* __restrict__ wt,
    const float* __restrict__ bqv, const float* __restrict__ bkv,
    const float* __restrict__ bvv,
    _Float16* __restrict__ qo, _Float16* __restrict__ ko,
    _Float16* __restrict__ vo) {
    const int lane = threadIdx.x;
    const int hi = lane >> 4, lr = lane & 15;
    const int r0 = blockIdx.x * 32, c0 = blockIdx.y * 64, which = blockIdx.z;

    const float* xrow0 = X + (size_t)(r0 + lr) * HIDq;
    const float* xrow1 = X + (size_t)(r0 + 16 + lr) * HIDq;
    const _Float16* wbase = wt + (size_t)which * HIDq * HIDq;
    const float*    bias  = (which == 0) ? bqv : ((which == 1) ? bkv : bvv);

    v8f acc[2][4] = {{{}, {}, {}, {}}, {{}, {}, {}, {}}};
    for (int k0 = 0; k0 < HIDq; k0 += 32) {
        AF a0, a1;
        const float* p0 = xrow0 + k0 + 16 * hi;
        const float* p1 = xrow1 + k0 + 16 * hi;
#pragma unroll
        for (int i = 0; i < 8; ++i) { a0.h[i] = (_Float16)p0[i];     a1.h[i] = (_Float16)p1[i]; }
#pragma unroll
        for (int i = 0; i < 8; ++i) { a0.h[8 + i] = (_Float16)p0[8 + i];
                                      a1.h[8 + i] = (_Float16)p1[8 + i]; }
#pragma unroll
        for (int t = 0; t < 4; ++t) {
            v16h bf = load_bfrag(wbase + (size_t)(c0 + t * 16 + lr) * HIDq + k0 + 16 * hi);
            acc[0][t] = wmma_f16(a0.v, bf, acc[0][t]);
            acc[1][t] = wmma_f16(a1.v, bf, acc[1][t]);
        }
    }

#pragma unroll
    for (int t = 0; t < 4; ++t) {
        const int col = c0 + t * 16 + lr;
        const int h = col >> 6, d = col & 63;
        const float bb = bias[col];
#pragma unroll
        for (int rt = 0; rt < 2; ++rt)
#pragma unroll
            for (int j = 0; j < 8; ++j) {
                int row = r0 + rt * 16 + j + 8 * hi;     // global (b*s) row
                int bI = row >> 10, s = row & 1023;
                _Float16 val = (_Float16)(acc[rt][t][j] + bb);
                if (which == 2) {
                    vo[((size_t)(bI * Hq + h) * Dq + d) * Sq + s] = val;  // [B,H,D,S]
                } else {
                    _Float16* dst = (which == 0) ? qo : ko;
                    dst[((size_t)(bI * Hq + h) * Sq + s) * Dq + d] = val; // [B,H,S,D]
                }
            }
    }
}

// ---------------------------------------------------------------------------
// Kernel 2: dual-softmax gated attention. One wave per (b, h, 16-row q tile).
// LDS: 16 x 1048 f16 (33.5 KB -> ~9 workgroups/WGP). Scores staged f16,
// then overwritten in place with gate-combined probabilities so pass 2's
// A-fragments are two contiguous ds_load_b128 per chunk.
// ---------------------------------------------------------------------------
#define SSTRH 1048              // halves; 524 dwords = 12 mod 64 bank stagger
#define NEGBIG (-3.4028235e38f)

__global__ __launch_bounds__(32) void k_attn(
    const _Float16* __restrict__ q, const _Float16* __restrict__ kk,
    const _Float16* __restrict__ vt, const float* __restrict__ amask,
    const int* __restrict__ lmask, const float* __restrict__ gate,
    float* __restrict__ out) {
    extern __shared__ _Float16 smh[];   // 16 * SSTRH halves
    const int lane = threadIdx.x;
    const int hi = lane >> 4, lr = lane & 15;
    const int q0 = blockIdx.x * 16, h = blockIdx.y, b = blockIdx.z;

    const _Float16* qbase = q  + (size_t)(b * Hq + h) * Sq * Dq;
    const _Float16* kbase = kk + (size_t)(b * Hq + h) * Sq * Dq;
    const _Float16* vbase = vt + (size_t)(b * Hq + h) * Dq * Sq;

    // Hoisted Q A-fragments (d = 0..31 and d = 32..63)
    const _Float16* qrow = qbase + (size_t)(q0 + lr) * Dq;
    const v16h a0 = load_afrag(qrow, 0 + 16 * hi);
    const v16h a1 = load_afrag(qrow, 32 + 16 * hi);

    // ---- Pass 1: scores = (Q K^T) / 8  ->  LDS (f16) ------------------------
    for (int kt = 0; kt < Sq / 16; ++kt) {
        const _Float16* krow = kbase + (size_t)(kt * 16 + lr) * Dq;
        v16h b0 = load_bfrag(krow + 0 + 16 * hi);
        v16h b1 = load_bfrag(krow + 32 + 16 * hi);
        v8f c = {};
        c = wmma_f16(a0, b0, c);
        c = wmma_f16(a1, b1, c);
#pragma unroll
        for (int j = 0; j < 8; ++j)
            smh[(j + 8 * hi) * SSTRH + kt * 16 + lr] = (_Float16)(c[j] * 0.125f);
    }
    __syncthreads();

    // ---- Pass 1b: per-row online (max,sum), both paths ----------------------
    // Lane owns row lr; lane pair (l, l+16) splits k by parity so both halves
    // of a pair read the same LDS dword (broadcast, conflict-free).
    const float* amb  = amask + (size_t)b * Sq;
    const int*   lrow = lmask + ((size_t)b * Sq + (q0 + lr)) * Sq;
    __builtin_prefetch(lrow, 0, 1);              // global_prefetch_b8

    _Float16* srow = smh + lr * SSTRH;
    float mg = -INFINITY, sg = 0.f, ml = -INFINITY, sl = 0.f;
    for (int i = 0; i < Sq / 2; ++i) {
        int k = 2 * i + hi;
        float sc = (float)srow[k];
        float g = sc + amb[k];
        if (g > mg) { sg = sg * __expf(mg - g) + 1.f; mg = g; }
        else        { sg += __expf(g - mg); }
        float lv = lrow[k] ? sc : NEGBIG;
        if (lv > ml) { sl = sl * __expf(ml - lv) + 1.f; ml = lv; }
        else         { sl += __expf(lv - ml); }
    }
    {   // merge the parity halves (same row, disjoint k sets)
        float mo = __shfl_xor(mg, 16, 32), so = __shfl_xor(sg, 16, 32);
        float M = fmaxf(mg, mo);
        sg = sg * __expf(mg - M) + so * __expf(mo - M); mg = M;
        mo = __shfl_xor(ml, 16, 32); so = __shfl_xor(sl, 16, 32);
        M = fmaxf(ml, mo);
        sl = sl * __expf(ml - M) + so * __expf(mo - M); ml = M;
    }
    const float gv = gate[(size_t)(b * Hq + h) * Sq + q0 + lr];
    const float wl = (sl > 0.f) ? (gv / sl) : 0.f;
    const float wg = (sg > 0.f) ? ((1.f - gv) / sg) : 0.f;

    // ---- Pass 1c: overwrite scores in place with combined probabilities -----
    // p = gate*p_local + (1-gate)*p_global, folded per row (gate is per-row).
    for (int i = 0; i < Sq / 2; ++i) {
        int k = 2 * i + hi;
        float sc = (float)srow[k];
        float pg = __expf(sc + amb[k] - mg) * wg;
        float pl = lrow[k] ? (__expf(sc - ml) * wl) : 0.f;
        srow[k] = (_Float16)(pl + pg);
    }
    __syncthreads();

    // ---- Pass 2: ctx = P_combined @ V  (A-frags straight from LDS) ----------
    v8f acc[4] = {{}, {}, {}, {}};
    for (int kc = 0; kc < Sq; kc += 32) {
        v16h a = load_afrag(srow, kc + 16 * hi);   // two contiguous 16B LDS reads
#pragma unroll
        for (int t = 0; t < 4; ++t) {
            const _Float16* vrow = vbase + (size_t)(t * 16 + lr) * Sq;
            v16h bf = load_bfrag(vrow + kc + 16 * hi);
            acc[t] = wmma_f16(a, bf, acc[t]);
        }
    }

    // ---- Write [B,S,HID] ----------------------------------------------------
    float* obase = out + (size_t)b * Sq * HIDq + h * Dq;
#pragma unroll
    for (int t = 0; t < 4; ++t)
#pragma unroll
        for (int j = 0; j < 8; ++j) {
            int M = j + 8 * hi;
            obase[(size_t)(q0 + M) * HIDq + t * 16 + lr] = acc[t][j];
        }
}

// ---------------------------------------------------------------------------
extern "C" void kernel_launch(void* const* d_in, const int* in_sizes, int n_in,
                              void* d_out, int out_size, void* d_ws, size_t ws_size,
                              hipStream_t stream) {
    (void)in_sizes; (void)n_in; (void)out_size; (void)ws_size;
    const float* hidden = (const float*)d_in[0];
    const float* amask  = (const float*)d_in[1];
    const int*   lmask  = (const int*)d_in[2];
    const float* gate   = (const float*)d_in[3];
    const float* Wq     = (const float*)d_in[4];
    const float* bq     = (const float*)d_in[5];
    const float* Wk     = (const float*)d_in[6];
    const float* bk     = (const float*)d_in[7];
    const float* Wv     = (const float*)d_in[8];
    const float* bv     = (const float*)d_in[9];

    // Workspace carve-up (f16): wt 3*768*768, q/k/vt 4*12*1024*64 each (~22.4 MB)
    _Float16* wt = (_Float16*)d_ws;
    _Float16* qb = wt + (size_t)3 * HIDq * HIDq;
    _Float16* kb = qb + (size_t)Bq * Hq * Sq * Dq;
    _Float16* vb = kb + (size_t)Bq * Hq * Sq * Dq;

    k_wcvt<<<dim3((3 * HIDq * HIDq) / 256), 256, 0, stream>>>(Wq, Wk, Wv, wt);
    k_qkv<<<dim3((Bq * Sq) / 32, HIDq / 64, 3), 32, 0, stream>>>(
        hidden, wt, bq, bk, bv, qb, kb, vb);
    k_attn<<<dim3(Sq / 16, Hq, Bq), 32, 16 * SSTRH * sizeof(_Float16), stream>>>(
        qb, kb, vb, amask, lmask, gate, (float*)d_out);
}
